// Evaluator_30047591202946
// MI455X (gfx1250) — matrix operand order repaired and checked by reading, not compiled, not run on hardware
//
#include <hip/hip_runtime.h>
#include <hip/hip_bf16.h>
#include <math.h>

// ---------------- types ----------------
typedef __bf16 bf16_t;
typedef bf16_t v16bf __attribute__((ext_vector_type(16)));
typedef bf16_t v8bf  __attribute__((ext_vector_type(8)));
typedef float  v8f   __attribute__((ext_vector_type(8)));

typedef int v4i __attribute__((vector_size(16)));
typedef __attribute__((address_space(1))) v4i g_v4i;   // global int4
typedef __attribute__((address_space(3))) v4i l_v4i;   // LDS int4

#define NHEADS_ 8
#define ATT_D   512
#define NCH     256
#define NBINS   32
#define NNODES  4096
#define NEDGES  32768
#define NPAIRS  131072
#define DHEAD   64

#if __has_builtin(__builtin_amdgcn_global_load_async_to_lds_b128)
#define HAS_ASYNC 1
#else
#define HAS_ASYNC 0
#endif

__device__ __forceinline__ bf16_t f2bf(float f) {
  unsigned u = __float_as_uint(f);
  unsigned r = u + 0x7FFFu + ((u >> 16) & 1u);   // round-to-nearest-even
  unsigned short h = (unsigned short)(r >> 16);
  return __builtin_bit_cast(bf16_t, h);
}

__device__ __forceinline__ void atomicMaxF(float* addr, float val) {
  unsigned* ua = (unsigned*)addr;
  unsigned old = *ua;
  while (__uint_as_float(old) < val) {
    unsigned assumed = old;
    old = atomicCAS(ua, assumed, __float_as_uint(val));
    if (old == assumed) break;
  }
}

// 16-byte global->LDS copy: async DMA when available, sync vector copy otherwise.
__device__ __forceinline__ void cp16(const bf16_t* g, bf16_t* l) {
#if HAS_ASYNC
  __builtin_amdgcn_global_load_async_to_lds_b128((g_v4i*)g, (l_v4i*)l, 0, 0);
#else
  *(v8bf*)l = *(const v8bf*)g;
#endif
}
__device__ __forceinline__ void cp_wait() {
#if HAS_ASYNC
#if __has_builtin(__builtin_amdgcn_s_wait_asynccnt)
  __builtin_amdgcn_s_wait_asynccnt(0);
#else
  asm volatile("s_wait_asynccnt 0x0" ::: "memory");
#endif
#endif
}

// ---------------- WMMA GEMM ----------------
// C[M,N] = act( concat(A0,A1)[M,K](bf16) @ Wt^T + bias (+ res) ), optional bf16 shadow of C.
// Wt: bf16 [Npad][K] row-major, Npad multiple of 128, rows >= N zero-filled.
// Requires: M % 128 == 0, K % 32 == 0.
#define BM 128
#define BN 128
#define BKK 32
#define BKP 40   // padded LDS row stride in bf16 (80B): keeps 16B chunks aligned

template<int ACT, bool RES, bool WBF>   // ACT: 0=none, 1=relu, 2=sigmoid
__global__ __launch_bounds__(256)
void k_gemm(const bf16_t* __restrict__ A0, const bf16_t* __restrict__ A1, int K0,
            const bf16_t* __restrict__ Wt, const float* __restrict__ bias,
            const float* __restrict__ res, float* __restrict__ C,
            bf16_t* __restrict__ Cbf, int M, int N, int K)
{
  __shared__ bf16_t sA[2][BM * BKP];
  __shared__ bf16_t sB[2][BN * BKP];
  const int tid  = threadIdx.x;
  const int lane = tid & 31;
  const int wid  = tid >> 5;     // 0..7
  const int half = lane >> 4;    // 0/1
  const int l16  = lane & 15;
  const int wm   = wid >> 1;     // 0..3  -> 32-row slab
  const int wn   = wid & 1;      // 0..1  -> 64-col slab
  const int m0   = blockIdx.y * BM;
  const int n0   = blockIdx.x * BN;
  const int K1   = K - K0;

  // per-thread staging chunks (2 for A, 2 for B; 16B each)
  const int idx0 = tid, idx1 = tid + 256;
  const int r0 = idx0 >> 2, c0 = (idx0 & 3) * 8;
  const int r1 = idx1 >> 2, c1 = (idx1 & 3) * 8;

  auto stage = [&](int buf, int kt) {
    {
      int gk = kt + c0;
      const bf16_t* g = (gk < K0) ? (A0 + (size_t)(m0 + r0) * K0 + gk)
                                  : (A1 + (size_t)(m0 + r0) * K1 + (gk - K0));
      cp16(g, &sA[buf][r0 * BKP + c0]);
      cp16(Wt + (size_t)(n0 + r0) * K + kt + c0, &sB[buf][r0 * BKP + c0]);
    }
    {
      int gk = kt + c1;
      const bf16_t* g = (gk < K0) ? (A0 + (size_t)(m0 + r1) * K0 + gk)
                                  : (A1 + (size_t)(m0 + r1) * K1 + (gk - K0));
      cp16(g, &sA[buf][r1 * BKP + c1]);
      cp16(Wt + (size_t)(n0 + r1) * K + kt + c1, &sB[buf][r1 * BKP + c1]);
    }
  };

  v8f acc[2][4];
#pragma unroll
  for (int mi = 0; mi < 2; ++mi)
#pragma unroll
    for (int ni = 0; ni < 4; ++ni)
      acc[mi][ni] = (v8f){0.f,0.f,0.f,0.f,0.f,0.f,0.f,0.f};

  stage(0, 0);
  int buf = 0;
  for (int kt = 0; kt < K; kt += BKK) {
    cp_wait();            // own async copies done
    __syncthreads();      // => all waves' copies done; prev buffer free
    if (kt + BKK < K) stage(buf ^ 1, kt + BKK);

    v16bf af[2], bfr[4];
#pragma unroll
    for (int mi = 0; mi < 2; ++mi) {
      // A layout: half 0 -> K 0..7 | 16..23 ; half 1 -> K 8..15 | 24..31
      const bf16_t* rp = &sA[buf][(wm * 32 + mi * 16 + l16) * BKP];
      v8bf lo = *(const v8bf*)(rp + 8 * half);
      v8bf hi = *(const v8bf*)(rp + 16 + 8 * half);
      af[mi] = __builtin_shufflevector(lo, hi, 0,1,2,3,4,5,6,7,8,9,10,11,12,13,14,15);
    }
#pragma unroll
    for (int ni = 0; ni < 4; ++ni) {
      // B layout: half 0 -> K 0..7 | 16..23 ; half 1 -> K 8..15 | 24..31 (column = l16)
      const bf16_t* rp = &sB[buf][(wn * 64 + ni * 16 + l16) * BKP];
      v8bf lo = *(const v8bf*)(rp + 8 * half);
      v8bf hi = *(const v8bf*)(rp + 16 + 8 * half);
      bfr[ni] = __builtin_shufflevector(lo, hi, 0,1,2,3,4,5,6,7,8,9,10,11,12,13,14,15);
    }
#pragma unroll
    for (int mi = 0; mi < 2; ++mi)
#pragma unroll
      for (int ni = 0; ni < 4; ++ni)
        acc[mi][ni] = __builtin_amdgcn_wmma_f32_16x16x32_bf16(
            false, af[mi], false, bfr[ni], (short)0, acc[mi][ni], false, false);
    buf ^= 1;
  }

  // epilogue: bias, residual, activation (+optional bf16 shadow)
#pragma unroll
  for (int mi = 0; mi < 2; ++mi)
#pragma unroll
    for (int ni = 0; ni < 4; ++ni) {
      int col = n0 + wn * 64 + ni * 16 + l16;
      if (col < N) {
        float bcol = bias[col];
#pragma unroll
        for (int i = 0; i < 8; ++i) {
          int row = m0 + wm * 32 + mi * 16 + half * 8 + i;
          float v = acc[mi][ni][i] + bcol;
          if (RES) v += res[(size_t)row * N + col];
          if (ACT == 1) v = fmaxf(v, 0.f);
          else if (ACT == 2) v = 1.f / (1.f + __expf(-v));
          C[(size_t)row * N + col] = v;
          if (WBF) Cbf[(size_t)row * N + col] = f2bf(v);
        }
      }
    }
}

// ---------------- helper kernels ----------------
__global__ void k_transw(const float* __restrict__ W, bf16_t* __restrict__ Wt,
                         int K, int N, int Npad) {
  int t = blockIdx.x * 256 + threadIdx.x;
  if (t >= Npad * K) return;
  int n = t / K, k = t - n * K;
  float v = (n < N) ? W[(size_t)k * N + n] : 0.f;   // [K][N] -> bf16 [Npad][K]
  Wt[t] = f2bf(v);
}

__global__ void k_fill(float* __restrict__ p, float v, int n) {
  int t = blockIdx.x * 256 + threadIdx.x;
  if (t < n) p[t] = v;
}

__global__ void k_b16(const float* __restrict__ s, bf16_t* __restrict__ d, int n) {
  int t = blockIdx.x * 256 + threadIdx.x;
  if (t < n) d[t] = f2bf(s[t]);
}

__global__ void k_initmem(const float* __restrict__ e, float* __restrict__ mem,
                          bf16_t* __restrict__ mem_bf, int n) {
  int t = blockIdx.x * 256 + threadIdx.x;
  if (t < n) { float v = e[t]; mem[t] = v; mem_bf[t] = f2bf(v); }
}

__global__ void k_embed(const float* __restrict__ x, const float* __restrict__ W,
                        const float* __restrict__ b, float* __restrict__ xh) {
  int t = blockIdx.x * 256 + threadIdx.x;
  if (t >= NNODES * NCH) return;
  int n = t >> 8, c = t & 255;
  const float* xp = x + n * 3;
  xh[t] = xp[0] * W[c] + xp[1] * W[NCH + c] + xp[2] * W[2 * NCH + c] + b[c];
}

__global__ void k_build(const float* __restrict__ xh, const float* __restrict__ edge,
                        const int* __restrict__ in, const int* __restrict__ jn,
                        bf16_t* __restrict__ ein_bf) {
  int t = blockIdx.x * 256 + threadIdx.x;
  if (t >= NEDGES * 1024) return;
  int e = t >> 10, kx = t & 1023;
  float v;
  if (kx < NCH)            v = xh[(size_t)in[e] * NCH + kx];
  else if (kx < 2 * NCH)   v = xh[(size_t)jn[e] * NCH + (kx - NCH)];
  else                     v = edge[(size_t)e * ATT_D + (kx - 2 * NCH)];
  ein_bf[t] = f2bf(v);
}

__global__ void k_layernorm(const float* __restrict__ in, const float* __restrict__ g,
                            const float* __restrict__ b, float* __restrict__ out,
                            bf16_t* __restrict__ obf, int rows, int D) {
  int wid = threadIdx.x >> 5, lane = threadIdx.x & 31;
  int row = blockIdx.x * 8 + wid;
  if (row >= rows) return;
  const float* x = in + (size_t)row * D;
  float s = 0.f, s2 = 0.f;
  for (int i = lane; i < D; i += 32) { float v = x[i]; s += v; s2 += v * v; }
  for (int off = 16; off; off >>= 1) {
    s  += __shfl_xor(s,  off, 32);
    s2 += __shfl_xor(s2, off, 32);
  }
  float mu  = s / D;
  float var = s2 / D - mu * mu;
  float inv = rsqrtf(var + 1e-5f);
  float* o = out + (size_t)row * D;
  bf16_t* ob = obf + (size_t)row * D;
  for (int i = lane; i < D; i += 32) {
    float v = (x[i] - mu) * inv * g[i] + b[i];
    o[i] = v;
    ob[i] = f2bf(v);
  }
}

__global__ void k_scores(const float* __restrict__ q, const float* __restrict__ k,
                         const int* __restrict__ ii, const int* __restrict__ jj,
                         float* __restrict__ s) {
  int t = blockIdx.x * 256 + threadIdx.x;
  if (t >= NPAIRS * NHEADS_) return;
  int p = t >> 3, h = t & 7;
  const float4* qp = (const float4*)(q + (size_t)ii[p] * ATT_D + h * DHEAD);
  const float4* kp = (const float4*)(k + (size_t)jj[p] * ATT_D + h * DHEAD);
  float acc = 0.f;
#pragma unroll
  for (int d = 0; d < DHEAD / 4; ++d) {
    float4 a = qp[d], b = kp[d];
    acc += a.x * b.x + a.y * b.y + a.z * b.z + a.w * b.w;
  }
  s[t] = acc * 0.125f;   // 1/sqrt(64)
}

__global__ void k_segmax(const float* __restrict__ s, const int* __restrict__ seg,
                         float* __restrict__ m) {
  int t = blockIdx.x * 256 + threadIdx.x;
  if (t >= NPAIRS * NHEADS_) return;
  int p = t >> 3, h = t & 7;
  atomicMaxF(&m[(size_t)seg[p] * NHEADS_ + h], s[t]);
}

__global__ void k_expsum(const float* __restrict__ s, const int* __restrict__ seg,
                         const float* __restrict__ m, float* __restrict__ den,
                         float* __restrict__ ex) {
  int t = blockIdx.x * 256 + threadIdx.x;
  if (t >= NPAIRS * NHEADS_) return;
  int p = t >> 3, h = t & 7;
  float e = __expf(s[t] - m[(size_t)seg[p] * NHEADS_ + h]);
  ex[t] = e;
  atomicAdd(&den[(size_t)seg[p] * NHEADS_ + h], e);
}

__global__ void k_msg(const float* __restrict__ ex, const float* __restrict__ den,
                      const float* __restrict__ v, const int* __restrict__ ii,
                      const int* __restrict__ jj, float* __restrict__ msg) {
  int t = blockIdx.x * 256 + threadIdx.x;
  if (t >= NPAIRS * NHEADS_) return;
  int p = t >> 3, h = t & 7;
  int si = ii[p], sj = jj[p];
  float a = ex[t] / den[(size_t)si * NHEADS_ + h];
  const float* vp = v + (size_t)sj * ATT_D + h * DHEAD;
  float* mp = msg + (size_t)si * ATT_D + h * DHEAD;
#pragma unroll 4
  for (int d = 0; d < DHEAD; ++d) atomicAdd(&mp[d], a * vp[d]);
}

__global__ void k_ufo(const float* __restrict__ g, float* __restrict__ e,
                      float* __restrict__ mem, bf16_t* __restrict__ mem_bf, int n) {
  int t = blockIdx.x * 256 + threadIdx.x;
  if (t >= n) return;
  float gg = g[t];
  float nv = gg * e[t] + (1.f - gg) * mem[t];
  e[t] = nv;
  mem[t] = nv;
  mem_bf[t] = f2bf(nv);
}

__global__ void k_nodesum(const float* __restrict__ e, const int* __restrict__ in,
                          float* __restrict__ node) {
  int t = blockIdx.x * 256 + threadIdx.x;
  if (t >= NEDGES * ATT_D) return;
  int ei = t >> 9, d = t & 511;
  atomicAdd(&node[(size_t)in[ei] * ATT_D + d], e[t]);
}

__global__ void k_outhead(const float* __restrict__ logits, const float* __restrict__ bins,
                          float* __restrict__ prob, float* __restrict__ err) {
  int t = blockIdx.x * 256 + threadIdx.x;
  if (t >= NNODES * 3) return;
  int n = t / 3, a = t - n * 3;
  const float* lg = logits + (size_t)n * 96 + a * NBINS;
  float mx = -3.0e38f;
#pragma unroll
  for (int b = 0; b < NBINS; ++b) mx = fmaxf(mx, lg[b]);
  float ebuf[NBINS]; float s = 0.f;
#pragma unroll
  for (int b = 0; b < NBINS; ++b) { ebuf[b] = __expf(lg[b] - mx); s += ebuf[b]; }
  float inv = 1.f / s, ep = 0.f;
  float* pr = prob + (size_t)n * 96 + a * NBINS;
#pragma unroll
  for (int b = 0; b < NBINS; ++b) { float pv = ebuf[b] * inv; pr[b] = pv; ep += pv * bins[b]; }
  err[t] = ep;
}

// ---------------- host side ----------------
static inline int cdiv(int a, int b) { return (a + b - 1) / b; }

static void gemm(hipStream_t st, const bf16_t* A0, const bf16_t* A1, int K0,
                 const bf16_t* Wt, const float* bias, const float* res,
                 float* C, bf16_t* Cbf, int M, int N, int K, int act) {
  dim3 g((unsigned)cdiv(N, BN), (unsigned)cdiv(M, BM)), b(256);
  if (!A1) { A1 = A0; K0 = K; }
#define GO(A_, R_, W_) k_gemm<A_, R_, W_><<<g, b, 0, st>>>(A0, A1, K0, Wt, bias, res, C, Cbf, M, N, K)
  if (res) {
    if (Cbf) { if (act == 1) GO(1,true,true);  else if (act == 2) GO(2,true,true);  else GO(0,true,true); }
    else     { if (act == 1) GO(1,true,false); else if (act == 2) GO(2,true,false); else GO(0,true,false); }
  } else {
    if (Cbf) { if (act == 1) GO(1,false,true);  else if (act == 2) GO(2,false,true);  else GO(0,false,true); }
    else     { if (act == 1) GO(1,false,false); else if (act == 2) GO(2,false,false); else GO(0,false,false); }
  }
#undef GO
}

// input indices (jax pytree order: dicts sorted by key, lists in order)
enum {
  IN_X = 0, IN_EDGE, IN_INODE, IN_JNODE, IN_IDXI, IN_IDXJ,
  IN_ATT0 = 6, IN_ATT1 = 22,
  IN_BINS = 38, IN_LN1B, IN_LN1G, IN_LXW, IN_LXB,
  IN_OUTW1, IN_OUTW2, IN_OUTB1, IN_OUTB2,
  IN_PREW1, IN_PREW2, IN_PREB1, IN_PREB2, IN_PREBT, IN_PREG,
  IN_UFOWG, IN_UFOBG
};
// per-att leaf offsets: W1,W2,Wk,Wo,Wq,Wv,b1,b2,b2n,bk,bo,bq,bv,g2,ln_b,ln_g
enum { AW1=0, AW2, AWK, AWO, AWQ, AWV, AB1, AB2, AB2N, ABK, ABO, ABQ, ABV, AG2, ALNB, ALNG };

extern "C" void kernel_launch(void* const* d_in, const int* in_sizes, int n_in,
                              void* d_out, int out_size, void* d_ws, size_t ws_size,
                              hipStream_t stream) {
  (void)in_sizes; (void)n_in; (void)out_size; (void)ws_size;
  const float* x      = (const float*)d_in[IN_X];
  const float* edge   = (const float*)d_in[IN_EDGE];
  const int*   i_node = (const int*)d_in[IN_INODE];
  const int*   j_node = (const int*)d_in[IN_JNODE];
  const int*   idx_i  = (const int*)d_in[IN_IDXI];
  const int*   idx_j  = (const int*)d_in[IN_IDXJ];
  const float* bins   = (const float*)d_in[IN_BINS];

  auto P = [&](int i) { return (const float*)d_in[i]; };

  // ---- workspace carve ----
  char* wp = (char*)d_ws;
  auto alloc = [&](size_t bytes) -> void* {
    void* p = (void*)wp;
    wp += (bytes + 255) & ~(size_t)255;
    return p;
  };
  const size_t ED = (size_t)NEDGES * ATT_D;
  float*  e      = (float*)alloc(ED * 4);
  float*  mem    = (float*)alloc(ED * 4);
  float*  h      = (float*)alloc(ED * 4);
  float*  ybuf   = (float*)alloc(ED * 4);
  float*  zbuf   = (float*)alloc(ED * 4);
  float*  qb     = (float*)alloc(ED * 4);
  float*  kb     = (float*)alloc(ED * 4);
  float*  vb     = (float*)alloc(ED * 4);
  float*  msg    = (float*)alloc(ED * 4);
  bf16_t* e_bf   = (bf16_t*)alloc(ED * 2);
  bf16_t* mem_bf = (bf16_t*)alloc(ED * 2);
  bf16_t* h_bf   = (bf16_t*)alloc(ED * 2);
  bf16_t* y_bf   = (bf16_t*)alloc(ED * 2);
  bf16_t* msg_bf = (bf16_t*)alloc(ED * 2);
  bf16_t* ein_bf = (bf16_t*)alloc((size_t)NEDGES * 1024 * 2);
  float*  sS     = (float*)alloc((size_t)NPAIRS * NHEADS_ * 4);
  float*  sEx    = (float*)alloc((size_t)NPAIRS * NHEADS_ * 4);
  float*  mmax   = (float*)alloc((size_t)NEDGES * NHEADS_ * 4);
  float*  den    = (float*)alloc((size_t)NEDGES * NHEADS_ * 4);
  float*  xh     = (float*)alloc((size_t)NNODES * NCH * 4);
  float*  node   = (float*)alloc((size_t)NNODES * ATT_D * 4);
  float*  nln    = (float*)alloc((size_t)NNODES * ATT_D * 4);
  bf16_t* nln_bf = (bf16_t*)alloc((size_t)NNODES * ATT_D * 2);
  float*  y2     = (float*)alloc((size_t)NNODES * 96 * 4);
  bf16_t* y2_bf  = (bf16_t*)alloc((size_t)NNODES * 96 * 2);
  float*  lgts   = (float*)alloc((size_t)NNODES * 96 * 4);
  bf16_t* preW1t = (bf16_t*)alloc((size_t)1024 * 512 * 2);   // [N=512][K=1024]
  bf16_t* preW2t = (bf16_t*)alloc((size_t)512 * 512 * 2);
  bf16_t* ufoWgt = (bf16_t*)alloc((size_t)1024 * 512 * 2);
  bf16_t* attWt[2][6];
  for (int blk = 0; blk < 2; ++blk)
    for (int w = 0; w < 6; ++w)
      attWt[blk][w] = (bf16_t*)alloc((size_t)512 * 512 * 2);
  bf16_t* outW1t = (bf16_t*)alloc((size_t)128 * 512 * 2);    // padded to 128 rows
  bf16_t* outW2t = (bf16_t*)alloc((size_t)128 * 96 * 2);     // padded to 128 rows

  hipStream_t st = stream;
  dim3 b256(256);
  auto grid1 = [&](int n) { return dim3((unsigned)cdiv(n, 256)); };

  // ---- transpose+convert all weights to bf16 [Npad][K] ----
  k_transw<<<grid1(512 * 1024), b256, 0, st>>>(P(IN_PREW1), preW1t, 1024, 512, 512);
  k_transw<<<grid1(512 * 512),  b256, 0, st>>>(P(IN_PREW2), preW2t, 512, 512, 512);
  k_transw<<<grid1(512 * 1024), b256, 0, st>>>(P(IN_UFOWG), ufoWgt, 1024, 512, 512);
  const int wleaf[6] = { AWQ, AWK, AWV, AWO, AW1, AW2 };
  for (int blk = 0; blk < 2; ++blk) {
    int base = (blk == 0) ? IN_ATT0 : IN_ATT1;
    for (int w = 0; w < 6; ++w)
      k_transw<<<grid1(512 * 512), b256, 0, st>>>(P(base + wleaf[w]), attWt[blk][w], 512, 512, 512);
  }
  k_transw<<<grid1(128 * 512), b256, 0, st>>>(P(IN_OUTW1), outW1t, 512, 96, 128);
  k_transw<<<grid1(128 * 96),  b256, 0, st>>>(P(IN_OUTW2), outW2t, 96, 96, 128);

  // ---- node embed + edge feature concat (bf16 GEMM operand) ----
  k_embed<<<grid1(NNODES * NCH), b256, 0, st>>>(x, P(IN_LXW), P(IN_LXB), xh);
  k_build<<<grid1(NEDGES * 1024), b256, 0, st>>>(xh, edge, i_node, j_node, ein_bf);

  // ---- pre ResLayer (act2=relu, norm) ----
  gemm(st, ein_bf, nullptr, 0, preW1t, P(IN_PREB1), nullptr, ybuf, y_bf, NEDGES, 512, 1024, 1);
  gemm(st, y_bf, nullptr, 0, preW2t, P(IN_PREB2), ybuf, zbuf, nullptr, NEDGES, 512, 512, 1);
  k_layernorm<<<dim3((unsigned)cdiv(NEDGES, 8)), b256, 0, st>>>(zbuf, P(IN_PREG), P(IN_PREBT), e, e_bf, NEDGES, 512);
  k_initmem<<<grid1((int)ED), b256, 0, st>>>(e, mem, mem_bf, (int)ED);

  // ---- attention blocks ----
  const int PH = NPAIRS * NHEADS_;
  for (int blk = 0; blk < 2; ++blk) {
    int base = (blk == 0) ? IN_ATT0 : IN_ATT1;
    k_layernorm<<<dim3((unsigned)cdiv(NEDGES, 8)), b256, 0, st>>>(e, P(base + ALNG), P(base + ALNB), h, h_bf, NEDGES, 512);
    gemm(st, h_bf, nullptr, 0, attWt[blk][0], P(base + ABQ), nullptr, qb, nullptr, NEDGES, 512, 512, 0);
    gemm(st, h_bf, nullptr, 0, attWt[blk][1], P(base + ABK), nullptr, kb, nullptr, NEDGES, 512, 512, 0);
    gemm(st, h_bf, nullptr, 0, attWt[blk][2], P(base + ABV), nullptr, vb, nullptr, NEDGES, 512, 512, 0);
    k_scores<<<grid1(PH), b256, 0, st>>>(qb, kb, idx_i, idx_j, sS);
    k_fill<<<grid1(NEDGES * NHEADS_), b256, 0, st>>>(mmax, -3.0e38f, NEDGES * NHEADS_);
    k_segmax<<<grid1(PH), b256, 0, st>>>(sS, idx_i, mmax);
    k_fill<<<grid1(NEDGES * NHEADS_), b256, 0, st>>>(den, 0.f, NEDGES * NHEADS_);
    k_expsum<<<grid1(PH), b256, 0, st>>>(sS, idx_i, mmax, den, sEx);
    k_fill<<<grid1((int)ED), b256, 0, st>>>(msg, 0.f, (int)ED);
    k_msg<<<grid1(PH), b256, 0, st>>>(sEx, den, vb, idx_i, idx_j, msg);
    k_b16<<<grid1((int)ED), b256, 0, st>>>(msg, msg_bf, (int)ED);
    // e += msg @ Wo + bo ; also emit bf16(e) for the FFN input
    gemm(st, msg_bf, nullptr, 0, attWt[blk][3], P(base + ABO), e, e, e_bf, NEDGES, 512, 512, 0);
    gemm(st, e_bf, nullptr, 0, attWt[blk][4], P(base + AB1), nullptr, ybuf, y_bf, NEDGES, 512, 512, 1);
    gemm(st, y_bf, nullptr, 0, attWt[blk][5], P(base + AB2), e, zbuf, nullptr, NEDGES, 512, 512, 0);
    k_layernorm<<<dim3((unsigned)cdiv(NEDGES, 8)), b256, 0, st>>>(zbuf, P(base + AG2), P(base + AB2N), e, e_bf, NEDGES, 512);
    // UFO gate: g = sigmoid(concat(e, mem) @ Wg + bg)
    gemm(st, e_bf, mem_bf, 512, ufoWgt, P(IN_UFOBG), nullptr, ybuf, nullptr, NEDGES, 512, 1024, 2);
    k_ufo<<<grid1((int)ED), b256, 0, st>>>(ybuf, e, mem, mem_bf, (int)ED);
  }

  // ---- node aggregation + output head ----
  k_fill<<<grid1(NNODES * ATT_D), b256, 0, st>>>(node, 0.f, NNODES * ATT_D);
  k_nodesum<<<grid1((int)ED), b256, 0, st>>>(e, i_node, node);
  k_layernorm<<<dim3((unsigned)cdiv(NNODES, 8)), b256, 0, st>>>(node, P(IN_LN1G), P(IN_LN1B), nln, nln_bf, NNODES, 512);
  gemm(st, nln_bf, nullptr, 0, outW1t, P(IN_OUTB1), nullptr, y2, y2_bf, NNODES, 96, 512, 1);
  gemm(st, y2_bf, nullptr, 0, outW2t, P(IN_OUTB2), y2, lgts, nullptr, NNODES, 96, 96, 0);

  float* prob = (float*)d_out;
  float* err  = prob + (size_t)NNODES * 96;
  k_outhead<<<grid1(NNODES * 3), b256, 0, st>>>(lgts, bins, prob, err);
}
